// EarthAttention3D_54511724921589
// MI455X (gfx1250) — matrix operand order repaired and checked by reading, not compiled, not run on hardware
//
#include <hip/hip_runtime.h>

// ---------------------------------------------------------------------------
// Pangu EarthAttention3D (dim=192) for gfx1250 (MI455X), wave32 + WMMA bf16.
// Mixed precision: bf16 WMMA operands, fp32 accumulation, fp32 softmax/bias.
// ---------------------------------------------------------------------------

typedef __attribute__((ext_vector_type(16))) __bf16          v16bf;
typedef __attribute__((ext_vector_type(16))) unsigned short  v16u;
typedef __attribute__((ext_vector_type(8)))  float           v8f;

#define NW_    8
#define TW_    124
#define NTOK   144
#define DIM    192
#define HEADS  6
#define DH     32
#define NBW    (NW_ * TW_)            // 992 window-batches
#define NWAVES 9                      // 9 waves * 16 tokens = 144
#define BLK_T  (NWAVES * 32)          // 288 threads
#define VPAD   168                    // padded key stride: 336 B, 16B-aligned,
                                      // 84 dwords == 20 mod 64 banks
#define SCALE  0.1767766952966369f    // 32^-0.5

__device__ __forceinline__ unsigned short f2bf(float f) {
  // fp32 -> bf16 RNE via hardware convert (v_cvt_pk_bf16_f32 on gfx1250)
  __bf16 h = (__bf16)f;
  return __builtin_bit_cast(unsigned short, h);
}

__device__ __forceinline__ v8f wmma_bf16(v16u a, v16u b, v8f c) {
  return __builtin_amdgcn_wmma_f32_16x16x32_bf16(
      false, __builtin_bit_cast(v16bf, a),
      false, __builtin_bit_cast(v16bf, b),
      (short)0, c, false, false);
}

// ---------------------------------------------------------------------------
// Kernel 1: convert + transpose weights to bf16 column-major ([out_col][in_k])
// so WMMA B-fragments are 16 contiguous bf16 (32 B) per lane.
// ---------------------------------------------------------------------------
__global__ void prep_weights(const float* __restrict__ w_qkv,   // [192][576]
                             const float* __restrict__ w_out,   // [192][192]
                             unsigned short* __restrict__ wqkvT, // [576][192]
                             unsigned short* __restrict__ woutT) // [192][192]
{
  int idx = blockIdx.x * 256 + threadIdx.x;
  if (idx < DIM * 3 * DIM) {                 // 110592
    int k = idx / (3 * DIM);
    int n = idx % (3 * DIM);
    wqkvT[n * DIM + k] = f2bf(w_qkv[idx]);
  } else {
    int j = idx - DIM * 3 * DIM;
    if (j < DIM * DIM) {
      int k = j / DIM;
      int n = j % DIM;
      woutT[n * DIM + k] = f2bf(w_out[j]);
    }
  }
}

// ---------------------------------------------------------------------------
// Kernel 2: fused per-(bw, head) QKV projection + attention.
//   grid = 992*6 blocks, 288 threads (9 waves), wave w owns query tile w.
// ---------------------------------------------------------------------------
__global__ __launch_bounds__(BLK_T, 1)
void fused_attn(const float* __restrict__ x,            // [992][144][192]
                const float* __restrict__ mask,         // [992][144][144]
                const float* __restrict__ b_qkv,        // [576]
                const float* __restrict__ bias,         // [124][6][144][144]
                const unsigned short* __restrict__ wqkvT, // [576][192] bf16
                unsigned short* __restrict__ obuf)      // [992][144][192] bf16
{
  __shared__ __attribute__((aligned(16))) unsigned short q_lds[NTOK][DH];
  __shared__ __attribute__((aligned(16))) unsigned short k_lds[NTOK][DH];
  // V transposed: [dh][key], key padded to 168 so rows are 16B-aligned and
  // bank-staggered; keys 144..159 zeroed (160..167 never read).
  __shared__ __attribute__((aligned(16))) unsigned short vT_lds[DH][VPAD];
  __shared__ __attribute__((aligned(16))) unsigned short p_lds[NWAVES][16][32];

  const int bwh = blockIdx.x;
  const int bw  = bwh / HEADS;          // window-batch (b*124 + w)
  const int h   = bwh % HEADS;
  const int wofs = bw % TW_;            // "w" index for bias

  const int tid  = threadIdx.x;
  const int wave = tid >> 5;
  const int lane = tid & 31;
  const int ln   = lane & 15;
  const int hi   = lane >> 4;

  // zero V padding keys 144..159 across all 32 dh rows
  for (int i = tid; i < DH * 16; i += BLK_T)
    vT_lds[i >> 4][NTOK + (i & 15)] = 0;

  // ---------------- Phase 1: QKV projection for this head -----------------
  // A-fragments of x (fp32 -> bf16 on the fly). A layout (16-bit, 16x32):
  //   lane: m = ln ; slot i: k = hi*16 + i
  const float* xrow = x + ((size_t)bw * NTOK + wave * 16 + ln) * DIM;
  v16u a[6];
#pragma unroll
  for (int kk = 0; kk < 6; ++kk) {
    const int kb = kk * 32 + hi * 16;
#pragma unroll
    for (int i = 0; i < 8; ++i) {
      a[kk][i]     = f2bf(xrow[kb + i]);
      a[kk][8 + i] = f2bf(xrow[kb + 8 + i]);
    }
  }

#pragma unroll
  for (int nt = 0; nt < 6; ++nt) {
    const int m3      = nt >> 1;                          // 0=q 1=k 2=v
    const int colbase = m3 * DIM + h * DH + (nt & 1) * 16; // col in [0,576)
    v8f acc = {};
#pragma unroll
    for (int kk = 0; kk < 6; ++kk) {
      // B layout (16-bit, 32x16): lane: n = colbase+ln ; slot i: k = hi*16+i
      const unsigned short* brow =
          wqkvT + (size_t)(colbase + ln) * DIM + kk * 32 + hi * 16;
      v16u bf;
#pragma unroll
      for (int i = 0; i < 16; ++i) bf[i] = brow[i];
      acc = wmma_bf16(a[kk], bf, acc);
    }
    const float bv = b_qkv[colbase + ln];
    const int dcol = (nt & 1) * 16 + ln;
    if (m3 < 2) {
      unsigned short (*dst)[DH] = (m3 == 0) ? q_lds : k_lds;
      const float sc = (m3 == 0) ? SCALE : 1.0f;          // q scaled after bias
#pragma unroll
      for (int j = 0; j < 8; ++j) {                        // C: m = j + 8*hi
        const int tok = wave * 16 + j + hi * 8;
        dst[tok][dcol] = f2bf((acc[j] + bv) * sc);
      }
    } else {
#pragma unroll
      for (int j = 0; j < 8; ++j) {                        // V stored transposed
        const int tok = wave * 16 + j + hi * 8;
        vT_lds[dcol][tok] = f2bf(acc[j] + bv);
      }
    }
  }

  __syncthreads();

  // ---------------- Phase 2: S = QK^T + bias + mask ------------------------
  // Q A-fragment: dh = 32 = WMMA K exactly -> 1 WMMA per 16x16 score tile.
  v16u qa;
  {
    const unsigned short* qrow = &q_lds[wave * 16 + ln][0];
#pragma unroll
    for (int i = 0; i < 8; ++i) {
      qa[i]     = qrow[hi * 16 + i];
      qa[8 + i] = qrow[hi * 16 + 8 + i];
    }
  }

  const float* biasbase = bias + (size_t)(wofs * HEADS + h) * NTOK * NTOK;
  const float* maskbase = mask + (size_t)bw * NTOK * NTOK;

  v8f s[9];
#pragma unroll
  for (int kt = 0; kt < 9; ++kt) {
    // B = K^T : lane: n(key) = kt*16+ln ; slot i: k(dh) = hi*16+i
    const unsigned short* krow = &k_lds[kt * 16 + ln][0] + hi * 16;
    v16u kb;
#pragma unroll
    for (int i = 0; i < 16; ++i) kb[i] = krow[i];
    v8f acc = {};
    acc = wmma_bf16(qa, kb, acc);
#pragma unroll
    for (int j = 0; j < 8; ++j) {
      const int nq = wave * 16 + j + hi * 8;   // query row
      const int mk = kt * 16 + ln;             // key col
      acc[j] += biasbase[(size_t)nq * NTOK + mk] +
                maskbase[(size_t)nq * NTOK + mk];
    }
    s[kt] = acc;
  }

  // ---------------- Phase 3: fp32 softmax over 144 keys --------------------
  // Row r = j + 8*hi lives in accumulator slot j across the 16 lanes of a
  // half-wave and across the 9 tiles -> tile-reduce then 16-lane xor-reduce.
  float inv[8];
#pragma unroll
  for (int j = 0; j < 8; ++j) {
    float m = s[0][j];
#pragma unroll
    for (int t = 1; t < 9; ++t) m = fmaxf(m, s[t][j]);
#pragma unroll
    for (int off = 1; off < 16; off <<= 1)
      m = fmaxf(m, __shfl_xor(m, off, 32));
    float sum = 0.0f;
#pragma unroll
    for (int t = 0; t < 9; ++t) {
      const float e = __expf(s[t][j] - m);
      s[t][j] = e;
      sum += e;
    }
#pragma unroll
    for (int off = 1; off < 16; off <<= 1)
      sum += __shfl_xor(sum, off, 32);
    inv[j] = 1.0f / sum;
  }

  // ---------------- Phase 4: O = P @ V (keys chunked by 32) ----------------
  v8f o0 = {}, o1 = {};
#pragma unroll
  for (int c = 0; c < 5; ++c) {
    const int t0 = 2 * c;
    const int t1 = (c < 4) ? 2 * c + 1 : 8;   // index kept in-bounds
    // stage P chunk (C layout -> [query row][key col] bf16); same-wave LDS
    // ops are in-order, so no barrier needed before re-reading.
#pragma unroll
    for (int j = 0; j < 8; ++j) {
      const int row = j + hi * 8;
      p_lds[wave][row][ln] = f2bf(s[t0][j] * inv[j]);
      p_lds[wave][row][16 + ln] =
          (c < 4) ? f2bf(s[t1][j] * inv[j]) : (unsigned short)0;
    }
    // A-fragment of P: lane: m = ln ; slot i: k = hi*16 + i
    v16u pa;
    const unsigned short* prow = &p_lds[wave][ln][0];
#pragma unroll
    for (int i = 0; i < 8; ++i) {
      pa[i]     = prow[hi * 16 + i];
      pa[8 + i] = prow[hi * 16 + 8 + i];
    }
    // B-fragments of V: lane: n(dh) = ln / 16+ln ; slot i: k = c*32+hi*16+i
    // vT layout makes each lane's 16 elements contiguous (32 B -> b128 loads)
    const unsigned short* vrow0 = &vT_lds[ln][c * 32 + hi * 16];
    const unsigned short* vrow1 = &vT_lds[16 + ln][c * 32 + hi * 16];
    v16u vb0, vb1;
#pragma unroll
    for (int i = 0; i < 16; ++i) {
      vb0[i] = vrow0[i];
      vb1[i] = vrow1[i];
    }
    o0 = wmma_bf16(pa, vb0, o0);
    o1 = wmma_bf16(pa, vb1, o1);
  }

  // store O (bf16) to [bw][tok][h*32 + dh]
  unsigned short* ob = obuf + (size_t)bw * NTOK * DIM + h * DH;
#pragma unroll
  for (int j = 0; j < 8; ++j) {
    const int tok = wave * 16 + j + hi * 8;
    ob[(size_t)tok * DIM + ln]      = f2bf(o0[j]);
    ob[(size_t)tok * DIM + 16 + ln] = f2bf(o1[j]);
  }
}

// ---------------------------------------------------------------------------
// Kernel 3: out = O @ w_out + b_out (fp32 result). 992 blocks x 9 waves.
// ---------------------------------------------------------------------------
__global__ __launch_bounds__(BLK_T, 1)
void out_proj(const unsigned short* __restrict__ obuf,   // [992][144][192] bf16
              const unsigned short* __restrict__ woutT,  // [192][192] bf16
              const float* __restrict__ b_out,           // [192]
              float* __restrict__ out)                   // [992][144][192] f32
{
  const int bw   = blockIdx.x;
  const int tid  = threadIdx.x;
  const int wave = tid >> 5;
  const int lane = tid & 31;
  const int ln   = lane & 15;
  const int hi   = lane >> 4;

  const unsigned short* orow =
      obuf + ((size_t)bw * NTOK + wave * 16 + ln) * DIM;
  v16u a[6];
#pragma unroll
  for (int kk = 0; kk < 6; ++kk) {
    const int kb = kk * 32 + hi * 16;
#pragma unroll
    for (int i = 0; i < 8; ++i) {
      a[kk][i]     = orow[kb + i];
      a[kk][8 + i] = orow[kb + 8 + i];
    }
  }

#pragma unroll
  for (int nt = 0; nt < 12; ++nt) {
    v8f acc = {};
#pragma unroll
    for (int kk = 0; kk < 6; ++kk) {
      const unsigned short* brow =
          woutT + (size_t)(nt * 16 + ln) * DIM + kk * 32 + hi * 16;
      v16u bf;
#pragma unroll
      for (int i = 0; i < 16; ++i) bf[i] = brow[i];
      acc = wmma_bf16(a[kk], bf, acc);
    }
    const float bv = b_out[nt * 16 + ln];
#pragma unroll
    for (int j = 0; j < 8; ++j) {
      const int tok = wave * 16 + j + hi * 8;
      out[((size_t)bw * NTOK + tok) * DIM + nt * 16 + ln] = acc[j] + bv;
    }
  }
}

// ---------------------------------------------------------------------------
// Launch
// ---------------------------------------------------------------------------
extern "C" void kernel_launch(void* const* d_in, const int* in_sizes, int n_in,
                              void* d_out, int out_size, void* d_ws,
                              size_t ws_size, hipStream_t stream) {
  const float* x     = (const float*)d_in[0];
  const float* mask  = (const float*)d_in[1];
  const float* w_qkv = (const float*)d_in[2];
  const float* b_qkv = (const float*)d_in[3];
  const float* bias  = (const float*)d_in[4];
  const float* w_out = (const float*)d_in[5];
  const float* b_out = (const float*)d_in[6];
  float* out = (float*)d_out;

  // workspace layout (needs ~55.2 MB):
  //   [0)            wqkvT bf16 [576][192]   221184 B
  //   [221184)       woutT bf16 [192][192]    73728 B
  //   [294912)       obuf  bf16 [992][144][192]  54853632 B
  char* ws = (char*)d_ws;
  unsigned short* wqkvT = (unsigned short*)(ws);
  unsigned short* woutT = (unsigned short*)(ws + 221184);
  unsigned short* obuf  = (unsigned short*)(ws + 294912);

  prep_weights<<<576, 256, 0, stream>>>(w_qkv, w_out, wqkvT, woutT);
  fused_attn<<<NBW * HEADS, BLK_T, 0, stream>>>(x, mask, b_qkv, bias, wqkvT,
                                                obuf);
  out_proj<<<NBW, BLK_T, 0, stream>>>(obuf, woutT, b_out, out);
}
